// GAP_LSTM_68762426409196
// MI455X (gfx1250) — matrix operand order repaired and checked by reading, not compiled, not run on hardware
//
#include <hip/hip_runtime.h>
#include <hip/hip_bf16.h>

typedef __attribute__((ext_vector_type(16))) _Float16 v16h;
typedef __attribute__((ext_vector_type(8)))  float    v8f;

#define NN 4096           // nodes
#define FF 32             // features
#define BB 4              // batch
#define CC 128            // BB*FF columns per state
#define TM 64             // output rows per block
#define KB 32             // K-step (matches wmma f16 K)
#define SZ (BB*NN*FF)     // 524288 elems per state tensor

// ---- f32 -> packed 2 x f16 in one dword ----
static __device__ __forceinline__ unsigned int pack2(float a, float b) {
    _Float16 ha = (_Float16)a, hb = (_Float16)b;
    unsigned short ua = __builtin_bit_cast(unsigned short, ha);
    unsigned short ub = __builtin_bit_cast(unsigned short, hb);
    return (unsigned int)ua | ((unsigned int)ub << 16);
}

// ============================================================
// Aggregation GEMM:  O[n,c] = sum_k adj[n,k] * S[k,c]
// S viewed as [N, C=B*F]: column c -> (b=c/32, f=c%32),
// element (k,c) at S[(b*NN + k)*FF + f].
// Block tile: 64 (M) x 128 (C), K-loop step 32.
// 8 waves arranged 2(M) x 4(C); each wave: 32x32 = 4 wmma tiles.
// adj/state tiles converted f32->f16 into LDS; f32 accumulate.
// ============================================================
__global__ __launch_bounds__(256)
void agg_gemm(const float* __restrict__ adj,
              const float* __restrict__ S0, const float* __restrict__ S1,
              const float* __restrict__ S2,
              float* __restrict__ O0, float* __restrict__ O1,
              float* __restrict__ O2)
{
    const float* S = (blockIdx.y == 0) ? S0 : (blockIdx.y == 1 ? S1 : S2);
    float*       O = (blockIdx.y == 0) ? O0 : (blockIdx.y == 1 ? O1 : O2);

    // LDS tiles stored as dwords = 2 packed f16 along K (K-pairs)
    __shared__ unsigned int lA[TM * KB / 2];   // [row 64][kpair 16]  4 KB
    __shared__ unsigned int lB[CC * KB / 2];   // [col 128][kpair 16] 8 KB

    const int t    = threadIdx.x;
    const int lane = t & 31;
    const int w    = t >> 5;
    const int wm   = w & 1;       // M half within block tile
    const int wn   = w >> 1;      // C quarter
    const int m0   = blockIdx.x * TM;

    const bool hi = (lane >= 16);
    const int  lm = lane & 15;

    // A loader: thread -> row ar (0..63), 8 consecutive k at ac
    const int ar = t >> 2;
    const int ac = (t & 3) * 8;
    // B loader: thread -> column bc, 16 consecutive k at bkh
    const int bc  = t & 127;
    const int bb  = bc >> 5;
    const int bfk = bc & 31;
    const int bkh = (t >> 7) * 16;

    v8f acc[2][2];
    acc[0][0] = {}; acc[0][1] = {}; acc[1][0] = {}; acc[1][1] = {};

    for (int kk = 0; kk < NN; kk += KB) {
        __syncthreads();   // previous iteration's LDS reads complete

        // ---- stage A tile (adj rows m0..m0+63, cols kk..kk+31) ----
        {
            const float* p = adj + (size_t)(m0 + ar) * NN + kk + ac;
            float4 f0 = *(const float4*)p;
            float4 f1 = *(const float4*)(p + 4);
            unsigned int* q = &lA[ar * 16 + (ac >> 1)];
            q[0] = pack2(f0.x, f0.y); q[1] = pack2(f0.z, f0.w);
            q[2] = pack2(f1.x, f1.y); q[3] = pack2(f1.z, f1.w);
            if (kk + KB < NN)   // gfx1250 global_prefetch of next adj tile
                __builtin_prefetch(p + KB, 0, 1);
        }
        // ---- stage B tile (S rows kk..kk+31, all 128 columns) ----
        {
            const float* p = S + ((size_t)bb * NN + kk + bkh) * FF + bfk;
            unsigned int* q = &lB[bc * 16 + (bkh >> 1)];
#pragma unroll
            for (int i = 0; i < 8; i++) {
                float a = p[(2 * i) * FF];
                float b = p[(2 * i + 1) * FF];
                q[i] = pack2(a, b);
            }
        }
        __syncthreads();

        // ---- gather fragments per ISA 7.12.2 16-bit layouts ----
        union { unsigned int u[8]; v16h v; } afr[2], bfr[2];
#pragma unroll
        for (int mi = 0; mi < 2; mi++) {
            const int row = wm * 32 + mi * 16 + lm;   // A: M = lane%16
#pragma unroll
            for (int v = 0; v < 8; v++) {
                // lanes 0-15: K pairs {0..7} -> idx v, {16..23}-> v+4
                // lanes 16-31: {8..15} -> v+4, {24..31} -> v+8
                const int idxK = v + (v < 4 ? (hi ? 4 : 0) : (hi ? 8 : 4));
                afr[mi].u[v] = lA[row * 16 + idxK];
            }
        }
#pragma unroll
        for (int ni = 0; ni < 2; ni++) {
            const int c = wn * 32 + ni * 16 + lm;     // B: N = lane%16
#pragma unroll
            for (int v = 0; v < 8; v++)               // K = (hi?8:0)+(v<4?0:8)+2v+half
                bfr[ni].u[v] = lB[c * 16 + (hi ? 4 : 0) + v + (v < 4 ? 0 : 4)];
        }

#pragma unroll
        for (int mi = 0; mi < 2; mi++)
#pragma unroll
            for (int ni = 0; ni < 2; ni++)
                acc[mi][ni] = __builtin_amdgcn_wmma_f32_16x16x32_f16(
                    false, afr[mi].v, false, bfr[ni].v,
                    (short)0, acc[mi][ni], false, false);
    }

    // ---- store: D layout: VGPR v, lane l -> M = v + (l<16?0:8), N = l%16 ----
#pragma unroll
    for (int mi = 0; mi < 2; mi++)
#pragma unroll
        for (int ni = 0; ni < 2; ni++) {
            const int cg = wn * 32 + ni * 16 + lm;
            const int ob = cg >> 5, of = cg & 31;
#pragma unroll
            for (int v = 0; v < 8; v++) {
                const int row = m0 + wm * 32 + mi * 16 + v + (hi ? 8 : 0);
                O[((size_t)ob * NN + row) * FF + of] = acc[mi][ni][v];
            }
        }
}

// ============================================================
// Gate kernels: per-thread one (b,n) row; weights for one gate
// pair staged in LDS (2 x 32 x 64 f32 = 16 KB).
// ============================================================
static __device__ __forceinline__ float sigm(float x) {
    return 1.0f / (1.0f + __expf(-x));
}

static __device__ __forceinline__ void stage_pair(
    const float* __restrict__ W, const float* __restrict__ bias, int g0,
    float Ws[2][32][64], float bs[2][64], int t)
{
    const float4* src = (const float4*)(W + (size_t)g0 * 2048);
    float4* dst = (float4*)&Ws[0][0][0];
#pragma unroll
    for (int i = 0; i < 4; i++) dst[t + i * 256] = src[t + i * 256];
    if (t < 128) ((float*)bs)[t] = bias[g0 * 64 + t];
}

// combined pre-activation: _gcn(ax,Wg)[j] + _gcn(ah,Wg+1)[j]
static __device__ __forceinline__ float gate_pair(
    const float* axr, const float* ahr,
    const float (*Ws)[32][64], const float (*bs)[64], int j)
{
    float lsx = bs[0][j], rsx = bs[0][j + 32];
    float lsh = bs[1][j], rsh = bs[1][j + 32];
#pragma unroll
    for (int k = 0; k < 32; k++) {
        lsx += axr[k] * Ws[0][k][j];
        rsx += axr[k] * Ws[0][k][j + 32];
        lsh += ahr[k] * Ws[1][k][j];
        rsh += ahr[k] * Ws[1][k][j + 32];
    }
    return lsx * sigm(rsx) + lsh * sigm(rsh);
}

__global__ __launch_bounds__(256)
void gates1(const float* __restrict__ ax, const float* __restrict__ ah,
            const float* __restrict__ c_in,
            const float* __restrict__ W, const float* __restrict__ bias,
            float* __restrict__ c_out, float* __restrict__ h1_out)
{
    __shared__ float Ws[2][32][64];
    __shared__ float bs[2][64];
    const int t   = threadIdx.x;
    const int row = blockIdx.x * 256 + t;          // 0..16383 = b*N+n

    float axr[32], ahr[32];
    const float* pax = ax + (size_t)row * FF;
    const float* pah = ah + (size_t)row * FF;
#pragma unroll
    for (int k = 0; k < 32; k++) { axr[k] = pax[k]; ahr[k] = pah[k]; }

    float fg[32], ig[32], cn[32];

    stage_pair(W, bias, 0, Ws, bs, t); __syncthreads();
#pragma unroll 4
    for (int j = 0; j < 32; j++) fg[j] = sigm(gate_pair(axr, ahr, Ws, bs, j));
    __syncthreads();

    stage_pair(W, bias, 2, Ws, bs, t); __syncthreads();
#pragma unroll 4
    for (int j = 0; j < 32; j++) ig[j] = sigm(gate_pair(axr, ahr, Ws, bs, j));
    __syncthreads();

    stage_pair(W, bias, 4, Ws, bs, t); __syncthreads();
#pragma unroll 4
    for (int j = 0; j < 32; j++) {
        float tj = tanhf(gate_pair(axr, ahr, Ws, bs, j));
        cn[j] = fg[j] * c_in[(size_t)row * FF + j] + ig[j] * tj;
        c_out[(size_t)row * FF + j] = cn[j];
    }
    __syncthreads();

    stage_pair(W, bias, 6, Ws, bs, t); __syncthreads();
#pragma unroll 4
    for (int j = 0; j < 32; j++) {
        float oj = sigm(gate_pair(axr, ahr, Ws, bs, j));
        h1_out[(size_t)row * FF + j] = oj * tanhf(cn[j]);
    }
}

__global__ __launch_bounds__(256)
void gates2(const float* __restrict__ ah1, const float* __restrict__ am,
            const float* __restrict__ m_in,
            const float* __restrict__ W, const float* __restrict__ bias,
            float* __restrict__ h_out, float* __restrict__ m_out)
{
    __shared__ float Ws[2][32][64];
    __shared__ float bs[2][64];
    const int t   = threadIdx.x;
    const int row = blockIdx.x * 256 + t;

    float ahr1[32], amr[32];
    const float* ph = ah1 + (size_t)row * FF;
    const float* pm = am  + (size_t)row * FF;
#pragma unroll
    for (int k = 0; k < 32; k++) { ahr1[k] = ph[k]; amr[k] = pm[k]; }

    float i2[32], mn[32];

    stage_pair(W, bias, 8, Ws, bs, t); __syncthreads();
#pragma unroll 4
    for (int j = 0; j < 32; j++) i2[j] = sigm(gate_pair(ahr1, amr, Ws, bs, j));
    __syncthreads();

    stage_pair(W, bias, 10, Ws, bs, t); __syncthreads();
#pragma unroll 4
    for (int j = 0; j < 32; j++) {
        float g = sigm(gate_pair(ahr1, amr, Ws, bs, j));
        mn[j] = i2[j] * m_in[(size_t)row * FF + j] + (1.0f - i2[j]) * g;
        m_out[(size_t)row * FF + j] = mn[j];
    }
    __syncthreads();

    stage_pair(W, bias, 12, Ws, bs, t); __syncthreads();
#pragma unroll 4
    for (int j = 0; j < 32; j++) {
        float o2 = sigm(gate_pair(ahr1, amr, Ws, bs, j));
        h_out[(size_t)row * FF + j] = mn[j] * o2;
    }
}

// ============================================================
extern "C" void kernel_launch(void* const* d_in, const int* in_sizes, int n_in,
                              void* d_out, int out_size, void* d_ws, size_t ws_size,
                              hipStream_t stream)
{
    (void)in_sizes; (void)n_in; (void)out_size; (void)ws_size;

    const float* x    = (const float*)d_in[0];
    const float* h    = (const float*)d_in[1];
    const float* c    = (const float*)d_in[2];
    const float* m    = (const float*)d_in[3];
    const float* adj  = (const float*)d_in[4];
    const float* W    = (const float*)d_in[5];
    const float* bias = (const float*)d_in[6];

    float* out   = (float*)d_out;
    float* h_new = out;                  // [B,N,F]
    float* c_new = out + (size_t)SZ;     // [B,N,F]
    float* m_new = out + 2 * (size_t)SZ; // [B,N,F]

    float* ws  = (float*)d_ws;           // 5 * 2MB f32 scratch
    float* ax  = ws;
    float* ah  = ws + (size_t)SZ;
    float* am  = ws + 2 * (size_t)SZ;
    float* ah1 = ws + 3 * (size_t)SZ;
    float* h1  = ws + 4 * (size_t)SZ;

    // Phase 1: ax/ah/am = adj @ {x,h,m}
    agg_gemm<<<dim3(NN / TM, 3), 256, 0, stream>>>(adj, x, h, m, ax, ah, am);
    // Phase 2: LSTM gates -> c_new (output) and h1 (scratch)
    gates1<<<dim3(SZ / FF / 256), 256, 0, stream>>>(ax, ah, c, W, bias, c_new, h1);
    // Phase 3: ah1 = adj @ h1
    agg_gemm<<<dim3(NN / TM, 1), 256, 0, stream>>>(adj, h1, h1, h1, ah1, ah1, ah1);
    // Phase 4: memory-state gates -> m_new, h_new (outputs)
    gates2<<<dim3(SZ / FF / 256), 256, 0, stream>>>(ah1, am, m, W, bias, h_new, m_new);
}
